// HierS2Classifier_42245298323651
// MI455X (gfx1250) — matrix operand order repaired and checked by reading, not compiled, not run on hardware
//
#include <hip/hip_runtime.h>
#include <cstdint>

// CDNA5 / gfx1250 — wave32, v_wmma_f32_16x16x32_bf16, f32 via bf16 hi/lo split.
// All operands pre-converted into fragment-ordered bf16 hi/lo global buffers,
// so the GEMM inner loop is pure global_load_b128 -> v_wmma (no LDS, no VALU).

typedef __attribute__((ext_vector_type(16))) __bf16   v16bf;
typedef __attribute__((ext_vector_type(8)))  float    v8f;
typedef __attribute__((ext_vector_type(8)))  unsigned v8u;

#define NUM_LABELS_TOT 8646
#define BATCH_M        8192
#define MTILES         512      // 8192/16
#define ACT_KT         88       // ceil(2814/32)
#define X_KT           24       // 768/32

union Frag { v8u u; v16bf b; };

// RNE f32 -> bf16 split: v ~= hi + lo
__device__ __forceinline__ void split1(float v, unsigned& h, unsigned& l) {
  unsigned u = __float_as_uint(v);
  h = (u + 0x7FFFu + ((u >> 16) & 1u)) >> 16;
  float r = v - __uint_as_float(h << 16);
  unsigned w = __float_as_uint(r);
  l = (w + 0x7FFFu + ((w >> 16) & 1u)) >> 16;
}
__device__ __forceinline__ void split2_pack(float v0, float v1,
                                            unsigned& hiPack, unsigned& loPack) {
  unsigned h0, l0, h1, l1;
  split1(v0, h0, l0); split1(v1, h1, l1);
  hiPack = (h0 & 0xFFFFu) | (h1 << 16);
  loPack = (l0 & 0xFFFFu) | (l1 << 16);
}

// x -> xfrag (raw, level-0 A) and actfrag (relu(x), levels>=1 A), fragment order.
// One wave per 16x32 A-tile; lane's 8 dwords are its WMMA A-operand dwords.
__global__ __launch_bounds__(256)
void conv_x_frag(const float* __restrict__ x,
                 unsigned* __restrict__ xHi, unsigned* __restrict__ xLo,
                 unsigned* __restrict__ aHi, unsigned* __restrict__ aLo) {
  int wave = threadIdx.x >> 5, lane = threadIdx.x & 31;
  int tile = blockIdx.x * 8 + wave;        // 0 .. MTILES*X_KT-1
  int mtile = tile / X_KT, kt = tile - mtile * X_KT;
  int mm = lane & 15, grpL = lane >> 4;
  size_t xIdx = (size_t)tile * 256 + lane * 8;                       // dwords
  size_t aIdx = ((size_t)mtile * ACT_KT + kt) * 256 + lane * 8;
  int m = mtile * 16 + mm;
  #pragma unroll
  for (int dw = 0; dw < 8; ++dw) {
    int grp = grpL | ((dw >> 2) << 1);
    int k   = grp * 8 + (dw & 3) * 2;      // even
    float2 v = *(const float2*)&x[(size_t)m * 768 + kt * 32 + k];
    unsigned hp, lp;
    split2_pack(v.x, v.y, hp, lp);
    xHi[xIdx + dw] = hp; xLo[xIdx + dw] = lp;
    split2_pack(fmaxf(v.x, 0.f), fmaxf(v.y, 0.f), hp, lp);
    aHi[aIdx + dw] = hp; aLo[aIdx + dw] = lp;
  }
}

// W (row-major [Nout,K]) -> fragment-ordered B hi/lo, zero-padded to tiles.
__global__ __launch_bounds__(256)
void conv_w_frag(const float* __restrict__ W, int K, int Nout, int KT, int totTiles,
                 unsigned* __restrict__ wHi, unsigned* __restrict__ wLo) {
  int wave = threadIdx.x >> 5, lane = threadIdx.x & 31;
  int tile = blockIdx.x * 8 + wave;
  if (tile >= totTiles) return;
  int ntile = tile / KT, kt = tile - ntile * KT;
  int nn = lane & 15, grpL = lane >> 4;
  int n = ntile * 16 + nn;
  bool okN = n < Nout;
  int ns = okN ? n : 0;
  size_t idx = (size_t)tile * 256 + lane * 8;
  #pragma unroll
  for (int dw = 0; dw < 8; ++dw) {
    int k  = (grpL | ((dw >> 2) << 1)) * 8 + (dw & 3) * 2;           // even, B-fragment k for this dword
    int gk = kt * 32 + k;
    bool ok = okN && (gk < K);             // K is even -> pair validity
    int gks = (gk < K) ? gk : (K - 2);
    float2 v = *(const float2*)&W[(size_t)ns * K + gks];
    unsigned hp, lp;
    split2_pack(ok ? v.x : 0.f, ok ? v.y : 0.f, hp, lp);
    wHi[idx + dw] = hp; wLo[idx + dw] = lp;
  }
}

// Block: 128(M) x 64(N), 256 threads = 8 waves, each wave 32x32.
// A/B operands loaded directly from fragment-ordered global buffers.
__global__ __launch_bounds__(256)
void hier_gemm_wmma(const unsigned* __restrict__ Ahi, const unsigned* __restrict__ Alo,
                    int aKT, int K,
                    const unsigned* __restrict__ Bhi, const unsigned* __restrict__ Blo,
                    int bKT,
                    const float* __restrict__ bias, const int* __restrict__ labels,
                    int labOff, int Nout, float* __restrict__ logits,
                    unsigned short* __restrict__ actHiS, unsigned short* __restrict__ actLoS,
                    int actOff, int writeAct)
{
  const int tid  = threadIdx.x;
  const int lane = tid & 31;
  const int wave = tid >> 5;
  const int wm   = (wave >> 1) * 2;   // 0,2,4,6
  const int wn   = (wave & 1) * 2;    // 0,2
  const int mtile0 = blockIdx.y * 8;
  const int ntile0 = blockIdx.x * 4;

  v8f acc[2][2];
  const v8f vzero = {0.f, 0.f, 0.f, 0.f, 0.f, 0.f, 0.f, 0.f};
  acc[0][0] = vzero; acc[0][1] = vzero; acc[1][0] = vzero; acc[1][1] = vzero;

  const unsigned* pA0 = Ahi + ((size_t)(mtile0 + wm + 0) * aKT) * 256 + lane * 8;
  const unsigned* pA1 = Ahi + ((size_t)(mtile0 + wm + 1) * aKT) * 256 + lane * 8;
  const unsigned* qA0 = Alo + ((size_t)(mtile0 + wm + 0) * aKT) * 256 + lane * 8;
  const unsigned* qA1 = Alo + ((size_t)(mtile0 + wm + 1) * aKT) * 256 + lane * 8;
  const unsigned* pB0 = Bhi + ((size_t)(ntile0 + wn + 0) * bKT) * 256 + lane * 8;
  const unsigned* pB1 = Bhi + ((size_t)(ntile0 + wn + 1) * bKT) * 256 + lane * 8;
  const unsigned* qB0 = Blo + ((size_t)(ntile0 + wn + 0) * bKT) * 256 + lane * 8;
  const unsigned* qB1 = Blo + ((size_t)(ntile0 + wn + 1) * bKT) * 256 + lane * 8;

  const int KTfull = K >> 5;

  #pragma unroll 2
  for (int kt = 0; kt < KTfull; ++kt) {
    size_t o = (size_t)kt * 256;
    v16bf aH0 = *(const v16bf*)(pA0 + o), aL0 = *(const v16bf*)(qA0 + o);
    v16bf aH1 = *(const v16bf*)(pA1 + o), aL1 = *(const v16bf*)(qA1 + o);
    v16bf bH0 = *(const v16bf*)(pB0 + o), bL0 = *(const v16bf*)(qB0 + o);
    v16bf bH1 = *(const v16bf*)(pB1 + o), bL1 = *(const v16bf*)(qB1 + o);
    acc[0][0] = __builtin_amdgcn_wmma_f32_16x16x32_bf16(false, aH0, false, bH0, (short)0, acc[0][0], false, false);
    acc[0][0] = __builtin_amdgcn_wmma_f32_16x16x32_bf16(false, aH0, false, bL0, (short)0, acc[0][0], false, false);
    acc[0][0] = __builtin_amdgcn_wmma_f32_16x16x32_bf16(false, aL0, false, bH0, (short)0, acc[0][0], false, false);
    acc[0][1] = __builtin_amdgcn_wmma_f32_16x16x32_bf16(false, aH0, false, bH1, (short)0, acc[0][1], false, false);
    acc[0][1] = __builtin_amdgcn_wmma_f32_16x16x32_bf16(false, aH0, false, bL1, (short)0, acc[0][1], false, false);
    acc[0][1] = __builtin_amdgcn_wmma_f32_16x16x32_bf16(false, aL0, false, bH1, (short)0, acc[0][1], false, false);
    acc[1][0] = __builtin_amdgcn_wmma_f32_16x16x32_bf16(false, aH1, false, bH0, (short)0, acc[1][0], false, false);
    acc[1][0] = __builtin_amdgcn_wmma_f32_16x16x32_bf16(false, aH1, false, bL0, (short)0, acc[1][0], false, false);
    acc[1][0] = __builtin_amdgcn_wmma_f32_16x16x32_bf16(false, aL1, false, bH0, (short)0, acc[1][0], false, false);
    acc[1][1] = __builtin_amdgcn_wmma_f32_16x16x32_bf16(false, aH1, false, bH1, (short)0, acc[1][1], false, false);
    acc[1][1] = __builtin_amdgcn_wmma_f32_16x16x32_bf16(false, aH1, false, bL1, (short)0, acc[1][1], false, false);
    acc[1][1] = __builtin_amdgcn_wmma_f32_16x16x32_bf16(false, aL1, false, bH1, (short)0, acc[1][1], false, false);
  }

  // K tail (K%32 != 0): B is zero-padded; mask A dwords with k >= K (no exec change).
  if (K & 31) {
    int kt = KTfull;
    unsigned keep[8];
    #pragma unroll
    for (int dw = 0; dw < 8; ++dw) {
      int grp = (lane >> 4) | ((dw >> 2) << 1);
      int k   = kt * 32 + grp * 8 + (dw & 3) * 2;
      keep[dw] = (k < K) ? 0xFFFFFFFFu : 0u;
    }
    size_t o = (size_t)kt * 256;
    Frag aH0, aL0, aH1, aL1;
    aH0.u = *(const v8u*)(pA0 + o); aL0.u = *(const v8u*)(qA0 + o);
    aH1.u = *(const v8u*)(pA1 + o); aL1.u = *(const v8u*)(qA1 + o);
    #pragma unroll
    for (int dw = 0; dw < 8; ++dw) {
      aH0.u[dw] &= keep[dw]; aL0.u[dw] &= keep[dw];
      aH1.u[dw] &= keep[dw]; aL1.u[dw] &= keep[dw];
    }
    v16bf bH0 = *(const v16bf*)(pB0 + o), bL0 = *(const v16bf*)(qB0 + o);
    v16bf bH1 = *(const v16bf*)(pB1 + o), bL1 = *(const v16bf*)(qB1 + o);
    acc[0][0] = __builtin_amdgcn_wmma_f32_16x16x32_bf16(false, aH0.b, false, bH0, (short)0, acc[0][0], false, false);
    acc[0][0] = __builtin_amdgcn_wmma_f32_16x16x32_bf16(false, aH0.b, false, bL0, (short)0, acc[0][0], false, false);
    acc[0][0] = __builtin_amdgcn_wmma_f32_16x16x32_bf16(false, aL0.b, false, bH0, (short)0, acc[0][0], false, false);
    acc[0][1] = __builtin_amdgcn_wmma_f32_16x16x32_bf16(false, aH0.b, false, bH1, (short)0, acc[0][1], false, false);
    acc[0][1] = __builtin_amdgcn_wmma_f32_16x16x32_bf16(false, aH0.b, false, bL1, (short)0, acc[0][1], false, false);
    acc[0][1] = __builtin_amdgcn_wmma_f32_16x16x32_bf16(false, aL0.b, false, bH1, (short)0, acc[0][1], false, false);
    acc[1][0] = __builtin_amdgcn_wmma_f32_16x16x32_bf16(false, aH1.b, false, bH0, (short)0, acc[1][0], false, false);
    acc[1][0] = __builtin_amdgcn_wmma_f32_16x16x32_bf16(false, aH1.b, false, bL0, (short)0, acc[1][0], false, false);
    acc[1][0] = __builtin_amdgcn_wmma_f32_16x16x32_bf16(false, aL1.b, false, bH0, (short)0, acc[1][0], false, false);
    acc[1][1] = __builtin_amdgcn_wmma_f32_16x16x32_bf16(false, aH1.b, false, bH1, (short)0, acc[1][1], false, false);
    acc[1][1] = __builtin_amdgcn_wmma_f32_16x16x32_bf16(false, aH1.b, false, bL1, (short)0, acc[1][1], false, false);
    acc[1][1] = __builtin_amdgcn_wmma_f32_16x16x32_bf16(false, aL1.b, false, bH1, (short)0, acc[1][1], false, false);
  }

  // Epilogue: bias, permuted scatter to logits, relu(y) -> fragment-ordered act.
  const int rSel = lane >> 4;
  const int nl   = lane & 15;
  #pragma unroll
  for (int mt = 0; mt < 2; ++mt) {
    #pragma unroll
    for (int nt = 0; nt < 2; ++nt) {
      int n = (ntile0 + wn + nt) * 16 + nl;
      if (n < Nout) {
        int   lab = labels[labOff + n];
        float bv  = bias[n];
        float y[8];
        #pragma unroll
        for (int r = 0; r < 8; ++r) {
          y[r] = acc[mt][nt][r] + bv;
          int m = (mtile0 + wm + mt) * 16 + r + (rSel << 3);
          logits[(size_t)m * NUM_LABELS_TOT + lab] = y[r];
        }
        if (writeAct) {
          int c   = actOff + n;
          int ktc = c >> 5, kc = c & 31, grp = kc >> 3;
          int lnA = ((grp & 1) << 4);                       // + mm below
          int dw  = ((kc & 7) >> 1) + ((grp >> 1) << 2);
          int half = kc & 1;
          #pragma unroll
          for (int r = 0; r < 8; ++r) {
            int mm = r + (rSel << 3);
            size_t idx = (((size_t)(mtile0 + wm + mt) * ACT_KT + ktc) * 32 + (lnA + mm)) * 16
                         + dw * 2 + half;
            unsigned h, l;
            split1(fmaxf(y[r], 0.f), h, l);
            actHiS[idx] = (unsigned short)h;
            actLoS[idx] = (unsigned short)l;
          }
        }
      }
    }
  }
}

extern "C" void kernel_launch(void* const* d_in, const int* in_sizes, int n_in,
                              void* d_out, int out_size, void* d_ws, size_t ws_size,
                              hipStream_t stream) {
  const float* x = (const float*)d_in[0];
  const float* W[6]; const float* b[6];
  for (int i = 0; i < 6; ++i) {
    W[i] = (const float*)d_in[1 + 2 * i];
    b[i] = (const float*)d_in[2 + 2 * i];
  }
  const int* labels = (const int*)d_in[13];
  float* logits = (float*)d_out;

  const int ins[6]  = {768, 774, 798, 894, 1278, 2814};
  const int outs[6] = {6, 24, 96, 384, 1536, 6144};
  const int NT[6]   = {4, 4, 8, 24, 96, 384};     // padN/16 (padN = ceil(N/64)*64)
  const int KT[6]   = {24, 25, 25, 28, 40, 88};   // ceil(K/32)

  // Workspace layout (dword units; each 16x32 fragment tile = 256 dwords)
  unsigned* p = (unsigned*)d_ws;
  unsigned* actHi = p; p += (size_t)MTILES * ACT_KT * 256;
  unsigned* actLo = p; p += (size_t)MTILES * ACT_KT * 256;
  unsigned* xHi   = p; p += (size_t)MTILES * X_KT * 256;
  unsigned* xLo   = p; p += (size_t)MTILES * X_KT * 256;
  unsigned* wHi[6]; unsigned* wLo[6];
  for (int i = 0; i < 6; ++i) {
    wHi[i] = p; p += (size_t)NT[i] * KT[i] * 256;
    wLo[i] = p; p += (size_t)NT[i] * KT[i] * 256;
  }

  // Convert x (raw + relu) into fragment buffers: 512*24 tiles, 8 per block.
  conv_x_frag<<<dim3(MTILES * X_KT / 8), dim3(256), 0, stream>>>(x, xHi, xLo, actHi, actLo);
  // Convert weights (zero-padded to tile boundaries).
  for (int i = 0; i < 6; ++i) {
    int tot = NT[i] * KT[i];
    conv_w_frag<<<dim3((tot + 7) / 8), dim3(256), 0, stream>>>(
        W[i], ins[i], outs[i], KT[i], tot, wHi[i], wLo[i]);
  }

  int labOff = 0;
  for (int i = 0; i < 6; ++i) {
    const unsigned* Ah = (i == 0) ? xHi : actHi;
    const unsigned* Al = (i == 0) ? xLo : actLo;
    int aKT = (i == 0) ? X_KT : ACT_KT;
    dim3 grid(NT[i] / 4, BATCH_M / 128);
    int writeAct = (i < 5) ? 1 : 0;
    hier_gemm_wmma<<<grid, dim3(256), 0, stream>>>(
        Ah, Al, aKT, ins[i], wHi[i], wLo[i], KT[i],
        b[i], labels, labOff, outs[i], logits,
        (unsigned short*)actHi, (unsigned short*)actLo, ins[i], writeAct);
    labOff += outs[i];
  }
}
